// Self_Attn1_25546465477258
// MI455X (gfx1250) — matrix operand order repaired and checked by reading, not compiled, not run on hardware
//
#include <hip/hip_runtime.h>
#include <hip/hip_bf16.h>

// MI455X (gfx1250) fused self-attention.
// proj_kernel: x1 (exact fp32) + q/k/v via v_wmma_f32_16x16x32_bf16.
// attn_kernel: flash-attention (online softmax) fully in registers/LDS,
//              bf16 WMMA for QK^T and PV, ds_swizzle row reductions.

#define BATCH 4
#define CIN   64
#define COUT  64
#define NPIX  4096   // 64*64
#define DKQ   8      // dk = C_OUT/8

typedef __attribute__((ext_vector_type(16))) __bf16 v16bf;
typedef __attribute__((ext_vector_type(8)))  float  v8f;

union FragB16 {            // one 16-element bf16 WMMA fragment (A or B)
    uint4  q[2];
    __bf16 h[16];
    v16bf  v;
};
union Pack8 {              // 8 bf16 = 16 bytes
    uint4  q;
    __bf16 h[8];
};

__device__ __forceinline__ v8f wmma_bf16(v16bf a, v16bf b, v8f c) {
    return __builtin_amdgcn_wmma_f32_16x16x32_bf16(
        /*neg_a=*/false, a, /*neg_b=*/false, b,
        /*c_mod=*/(short)0, c, /*reuse_a=*/false, /*reuse_b=*/false);
}

template <int IMM>
__device__ __forceinline__ float swz(float v) {
    return __int_as_float(__builtin_amdgcn_ds_swizzle(__float_as_int(v), IMM));
}
// max-reduce across the 16 lanes of each wave32 half (xor 1,2,4,8)
__device__ __forceinline__ float rowmax16(float v) {
    v = fmaxf(v, swz<0x041F>(v));
    v = fmaxf(v, swz<0x081F>(v));
    v = fmaxf(v, swz<0x101F>(v));
    v = fmaxf(v, swz<0x201F>(v));
    return v;
}
// broadcast lane (16*half) value to all lanes of that half: and_mask=0x10
__device__ __forceinline__ float bcast_half(float v) { return swz<0x0010>(v); }

// convert 8 consecutive f32 -> bf16 fragment slots
__device__ __forceinline__ void cvt8(const float* __restrict__ p, __bf16* h) {
    float4 f0 = ((const float4*)p)[0];
    float4 f1 = ((const float4*)p)[1];
    h[0] = (__bf16)f0.x; h[1] = (__bf16)f0.y; h[2] = (__bf16)f0.z; h[3] = (__bf16)f0.w;
    h[4] = (__bf16)f1.x; h[5] = (__bf16)f1.y; h[6] = (__bf16)f1.z; h[7] = (__bf16)f1.w;
}
// A-fragment (16x32 bf16): lane row = l%16; elem e -> K = 32*ks + (e + (e>=8?8:0) + 8*half)
__device__ __forceinline__ v16bf loadA_w(const float* __restrict__ row, int ks, int half) {
    FragB16 a;
    int c0 = ks * 32 + half * 8;
    cvt8(row + c0,      &a.h[0]);
    cvt8(row + c0 + 16, &a.h[8]);
    return a.v;
}

// ---------------------------------------------------------------------------
// Kernel 1: projections. grid = B * (N/64) = 256 blocks, 128 threads (4 waves)
// ---------------------------------------------------------------------------
__global__ __launch_bounds__(128) void proj_kernel(
    const float* __restrict__ x,
    const float* __restrict__ Wq, const float* __restrict__ bq,
    const float* __restrict__ Wk, const float* __restrict__ bk,
    const float* __restrict__ Wv, const float* __restrict__ bv,
    const float* __restrict__ W1, const float* __restrict__ b1,
    float* __restrict__ out,
    __bf16* __restrict__ qT, __bf16* __restrict__ kT, __bf16* __restrict__ vbf)
{
    __shared__ float xs[64 * 64];                       // x tile, [c][n]
    __shared__ __align__(16) __bf16 xT[64 * 72];        // x tile transposed, [n][c]
    __shared__ float w1s[64 * 64];
    __shared__ float bqk[16];
    __shared__ float bvs[64];

    const int t    = threadIdx.x;
    const int b    = blockIdx.x >> 6;
    const int n0   = (blockIdx.x & 63) * 64;
    const int lane = t & 31;
    const int wid  = t >> 5;
    const int il   = lane & 15;
    const int half = lane >> 4;

    // ---- stage x (f32 + bf16-transposed), W1, biases
    {
        const int nl = t & 63, cg = t >> 6;
#pragma unroll 4
        for (int it = 0; it < 32; ++it) {
            int c = cg + it * 2;
            float v = x[((size_t)(b * CIN + c)) * NPIX + n0 + nl];
            xs[c * 64 + nl]   = v;
            xT[nl * 72 + c]   = (__bf16)v;
        }
#pragma unroll 4
        for (int it = 0; it < 32; ++it) w1s[t + it * 128] = W1[t + it * 128];
        if (t < 8)        bqk[t] = bq[t];
        else if (t < 16)  bqk[t] = bk[t - 8];
        if (t < 64)       bvs[t] = bv[t];
    }
    __syncthreads();

    // ---- x1 = W1 x + b1 in exact fp32 (this is the output when gamma==0)
    {
        const int nl = t & 63, og = t >> 6;
        for (int p = 0; p < 4; ++p) {
            int o0 = og * 32 + p * 8;
            float acc[8];
#pragma unroll
            for (int u = 0; u < 8; ++u) acc[u] = b1[o0 + u];
            for (int c = 0; c < 64; ++c) {
                float xv = xs[c * 64 + nl];
#pragma unroll
                for (int u = 0; u < 8; ++u)
                    acc[u] = fmaf(w1s[(o0 + u) * 64 + c], xv, acc[u]);
            }
#pragma unroll
            for (int u = 0; u < 8; ++u)
                out[((size_t)(b * COUT + o0 + u)) * NPIX + n0 + nl] = acc[u];
        }
    }

    // ---- q,k,v via bf16 WMMA. Each wave owns a 16-column n-subtile.
    {
        const int nloc = wid * 16 + il;                 // 0..63
        v8f qk = 0;
        v8f va0 = 0, va1 = 0, va2 = 0, va3 = 0;
#pragma unroll
        for (int ks = 0; ks < 2; ++ks) {
            // B fragment from xT: col n = nloc, elem e -> c = 32*ks + (e + (e>=8?8:0) + 8*half)
            FragB16 bx;
            const __bf16* xr = &xT[nloc * 72 + ks * 32 + half * 8];
            bx.q[0] = *(const uint4*)(xr);
            bx.q[1] = *(const uint4*)(xr + 16);

            const float* wqk_row = (il < 8) ? (Wq + il * CIN) : (Wk + (il - 8) * CIN);
            qk  = wmma_bf16(loadA_w(wqk_row, ks, half), bx.v, qk);
            va0 = wmma_bf16(loadA_w(Wv + (0 * 16 + il) * CIN, ks, half), bx.v, va0);
            va1 = wmma_bf16(loadA_w(Wv + (1 * 16 + il) * CIN, ks, half), bx.v, va1);
            va2 = wmma_bf16(loadA_w(Wv + (2 * 16 + il) * CIN, ks, half), bx.v, va2);
            va3 = wmma_bf16(loadA_w(Wv + (3 * 16 + il) * CIN, ks, half), bx.v, va3);
        }
        const int n = n0 + nloc;
        // qk tile: rows 0..7 = q channels (half0 lanes), rows 8..15 = k (half1)
        Pack8 pk;
#pragma unroll
        for (int r = 0; r < 8; ++r) pk.h[r] = (__bf16)(qk[r] + bqk[half * 8 + r]);
        __bf16* dst = (half == 0) ? (qT + ((size_t)b * NPIX + n) * DKQ)
                                  : (kT + ((size_t)b * NPIX + n) * DKQ);
        *(uint4*)dst = pk.q;
        // v tiles: element (c = ct*16 + r + 8*half, n)
        v8f vat[4] = {va0, va1, va2, va3};
#pragma unroll
        for (int ct = 0; ct < 4; ++ct)
#pragma unroll
            for (int r = 0; r < 8; ++r) {
                int c = ct * 16 + half * 8 + r;
                vbf[((size_t)(b * COUT + c)) * NPIX + n] = (__bf16)(vat[ct][r] + bvs[c]);
            }
    }
}

// ---------------------------------------------------------------------------
// Kernel 2: flash attention. grid = B * (N/64) = 256 blocks, 128 threads.
// Each wave owns one 16-row i-tile; loops over j in steps of 32.
// ---------------------------------------------------------------------------
__global__ __launch_bounds__(128) void attn_kernel(
    const __bf16* __restrict__ qT, const __bf16* __restrict__ kT,
    const __bf16* __restrict__ vbf, const float* __restrict__ gamma,
    float* __restrict__ out)
{
    __shared__ __align__(16) __bf16 ps[4][16 * 40];     // per-wave P transpose buffer

    const int t    = threadIdx.x;
    const int lane = t & 31;
    const int wid  = t >> 5;
    const int il   = lane & 15;
    const int half = lane >> 4;
    const int b    = blockIdx.x >> 6;
    const int itile = (blockIdx.x & 63) * 64 + wid * 16;
    __bf16* psw = ps[wid];

    const __bf16* qTb = qT  + (size_t)b * NPIX * DKQ;
    const __bf16* kTb = kT  + (size_t)b * NPIX * DKQ;
    const __bf16* vbb = vbf + (size_t)b * COUT * NPIX;

    const uint4 z4 = make_uint4(0u, 0u, 0u, 0u);

    // A_q fragment: row i = itile + il, K = dk channels (zero-padded to 32)
    FragB16 aq;
    aq.q[0] = z4; aq.q[1] = z4;
    if (half == 0) aq.q[0] = *(const uint4*)(qTb + (size_t)(itile + il) * DKQ);

    // ones-column B fragment (fuses softmax row-sum into an extra WMMA)
    FragB16 bones;
    bones.q[0] = z4; bones.q[1] = z4;
    if (il == 0) {
#pragma unroll
        for (int e = 0; e < 16; ++e) bones.h[e] = (__bf16)1.0f;
    }

    v8f O0 = 0, O1 = 0, O2 = 0, O3 = 0, Osum = 0;
    float m[8];
#pragma unroll
    for (int r = 0; r < 8; ++r) m[r] = -3.0e38f;

    for (int j0 = 0; j0 < NPIX; j0 += 32) {
        if (j0 + 32 < NPIX) {
            __builtin_prefetch(kTb + (size_t)(j0 + 32 + lane) * DKQ, 0, 1);
            __builtin_prefetch(vbb + (size_t)lane * NPIX + j0 + 32, 0, 1);
            __builtin_prefetch(vbb + (size_t)(32 + lane) * NPIX + j0 + 32, 0, 1);
        }
        // B_k fragments for j0 and j0+16 (dk zero-padded)
        FragB16 bk0, bk1;
        bk0.q[0] = z4; bk0.q[1] = z4; bk1.q[0] = z4; bk1.q[1] = z4;
        if (half == 0) {
            bk0.q[0] = *(const uint4*)(kTb + (size_t)(j0 + il) * DKQ);
            bk1.q[0] = *(const uint4*)(kTb + (size_t)(j0 + 16 + il) * DKQ);
        }
        v8f zz = 0;
        v8f S0 = wmma_bf16(aq.v, bk0.v, zz);
        v8f S1 = wmma_bf16(aq.v, bk1.v, zz);

        // online softmax: row max via ds_swizzle reductions
        float scale[8];
#pragma unroll
        for (int r = 0; r < 8; ++r) {
            float mx = rowmax16(fmaxf(S0[r], S1[r]));
            float mn = fmaxf(m[r], mx);
            scale[r] = __expf(m[r] - mn);
            m[r] = mn;
        }
        float P0[8], P1[8];
#pragma unroll
        for (int r = 0; r < 8; ++r) {
            P0[r] = __expf(S0[r] - m[r]);
            P1[r] = __expf(S1[r] - m[r]);
        }
#pragma unroll
        for (int r = 0; r < 8; ++r) {
            O0[r] *= scale[r]; O1[r] *= scale[r];
            O2[r] *= scale[r]; O3[r] *= scale[r];
            Osum[r] *= scale[r];
        }
        // P -> LDS (bf16, [i_local][j_local], stride 40 elems)
#pragma unroll
        for (int r = 0; r < 8; ++r) {
            int irow = r + half * 8;
            psw[irow * 40 + il]      = (__bf16)P0[r];
            psw[irow * 40 + 16 + il] = (__bf16)P1[r];
        }
        // A_p fragment: row i = il, K = j (same-wave LDS ops are in-order)
        FragB16 ap;
        {
            const __bf16* row = psw + il * 40 + half * 8;
            ap.q[0] = *(const uint4*)(row);
            ap.q[1] = *(const uint4*)(row + 16);
        }
        // PV: B = V^T fragment straight from j-contiguous bf16 V
#pragma unroll
        for (int ct = 0; ct < 4; ++ct) {
            FragB16 bv;
            const __bf16* vp = vbb + (size_t)(ct * 16 + il) * NPIX + j0 + half * 8;
            bv.q[0] = *(const uint4*)(vp);
            bv.q[1] = *(const uint4*)(vp + 16);
            v8f acc = (ct == 0) ? O0 : (ct == 1) ? O1 : (ct == 2) ? O2 : O3;
            acc = wmma_bf16(ap.v, bv.v, acc);
            if (ct == 0) O0 = acc; else if (ct == 1) O1 = acc;
            else if (ct == 2) O2 = acc; else O3 = acc;
        }
        Osum = wmma_bf16(ap.v, bones.v, Osum);
    }

    // epilogue: out = x1 + gamma/l * O  (out already holds x1)
    const float g = gamma[0];
#pragma unroll
    for (int r = 0; r < 8; ++r) {
        float lsum = bcast_half(Osum[r]);   // row sum lives in column 0 lanes
        float sc = g / lsum;
        int i = itile + r + half * 8;
        v8f Ot[4] = {O0, O1, O2, O3};
#pragma unroll
        for (int ct = 0; ct < 4; ++ct) {
            int c = ct * 16 + il;
            size_t oi = ((size_t)(b * COUT + c)) * NPIX + i;
            out[oi] += Ot[ct][r] * sc;
        }
    }
}

// ---------------------------------------------------------------------------
extern "C" void kernel_launch(void* const* d_in, const int* in_sizes, int n_in,
                              void* d_out, int out_size, void* d_ws, size_t ws_size,
                              hipStream_t stream) {
    const float* x     = (const float*)d_in[0];
    const float* Wq    = (const float*)d_in[1];
    const float* bq    = (const float*)d_in[2];
    const float* Wk    = (const float*)d_in[3];
    const float* bk    = (const float*)d_in[4];
    const float* Wv    = (const float*)d_in[5];
    const float* bv    = (const float*)d_in[6];
    const float* W1    = (const float*)d_in[7];
    const float* b1    = (const float*)d_in[8];
    const float* gamma = (const float*)d_in[9];
    float* out = (float*)d_out;

    // workspace layout (2.5 MB total): qT | kT | v(bf16)
    __bf16* qT  = (__bf16*)d_ws;
    __bf16* kT  = qT + (size_t)BATCH * NPIX * DKQ;
    __bf16* vbf = kT + (size_t)BATCH * NPIX * DKQ;

    dim3 grid(BATCH * (NPIX / 64));
    dim3 block(128);
    proj_kernel<<<grid, block, 0, stream>>>(x, Wq, bq, Wk, bk, Wv, bv, W1, b1,
                                            out, qT, kT, vbf);
    attn_kernel<<<grid, block, 0, stream>>>(qT, kT, vbf, gamma, out);
}
